// GroupedGEMM_12455405158653
// MI455X (gfx1250) — matrix operand order repaired and checked by reading, not compiled, not run on hardware
//
#include <hip/hip_runtime.h>

// ---------------------------------------------------------------------------
// Grouped GEMM for MI455X (gfx1250): fp32 in HBM -> bf16 WMMA (f32 accum).
// Fixed expert segmentation from the reference module constants.
// ---------------------------------------------------------------------------

constexpr int KD = 2560;   // in_features
constexpr int ND = 3328;   // out_features

constexpr int BM = 128, BN = 128, BK = 32;
constexpr int LDA = 40;    // bf16 elems per LDS row (32 + pad; 80B rows, 16B-aligned frags)
constexpr int LDB = 40;
constexpr int KSTEPS = KD / BK;   // 80

typedef __attribute__((ext_vector_type(16))) __bf16        v16bf;
typedef __attribute__((ext_vector_type(2)))  __bf16        v2bf;
typedef __attribute__((ext_vector_type(8)))  float         v8f;
typedef __attribute__((ext_vector_type(4)))  unsigned int  u32x4;
typedef __attribute__((ext_vector_type(2)))  unsigned int  u32x2;
typedef __attribute__((ext_vector_type(8)))  unsigned int  u32x8;

__device__ __forceinline__ unsigned int pack2_bf16(float a, float b) {
  // Native fptrunc f32->bf16 (RNE); lets the backend select v_cvt_pk_bf16_f32
  // instead of a manual integer rounding sequence.
  v2bf v;
  v.x = (__bf16)a;
  v.y = (__bf16)b;
  return __builtin_bit_cast(unsigned int, v);
}

__global__ void __launch_bounds__(256, 1)
grouped_gemm_wmma_bf16(const float* __restrict__ X,   // [4096, KD]; rows row0..row0+mg-1
                       const float* __restrict__ Wg,  // [KD, ND] expert slice
                       float* __restrict__ Out,       // [4096, ND]
                       int row0, int mg)
{
  __shared__ unsigned short As[2][BM * LDA];  // 2 x 10240 B
  __shared__ unsigned short Bs[2][BN * LDB];  // 2 x 10240 B (stored transposed: [n][k])

  const int tid   = threadIdx.x;       // 0..255
  const int lane  = tid & 31;
  const int wave  = tid >> 5;          // 0..7
  const int ln    = lane & 15;         // N / M index within 16x16 tile
  const int lh    = lane >> 4;         // lane half (K / M-half selector)
  const int waveM = (wave & 1) * 64;   // 2 wave rows
  const int waveN = (wave >> 1) * 32;  // 4 wave cols

  const int m0 = blockIdx.x * BM;
  const int n0 = blockIdx.y * BN;

  // B staging mapping: thread -> (column n, K half-block)
  const int b_n  = tid & 127;          // 0..127
  const int b_kh = tid >> 7;           // 0/1 -> K sub-block of 16

  float4 areg[4];
  float  breg[16];

  // ---- global -> registers for K-step ks --------------------------------
  auto load_tile = [&](int ks) {
    const int k0 = ks * BK;
    #pragma unroll
    for (int j = 0; j < 4; ++j) {
      int f   = tid + 256 * j;         // 0..1023 float4 slots
      int row = f >> 3;                // 0..127
      int kq  = (f & 7) * 4;           // 0,4,...,28
      int xr  = m0 + row;
      if (xr > mg - 1) xr = mg - 1;    // clamp: keep address valid, store-guard later
      areg[j] = *(const float4*)(X + (size_t)(row0 + xr) * KD + (k0 + kq));
    }
    const float* bp = Wg + (size_t)(k0 + b_kh * 16) * ND + (n0 + b_n);
    #pragma unroll
    for (int i = 0; i < 16; ++i)
      breg[i] = bp[(size_t)i * ND];    // coalesced across lanes at fixed k
    // prefetch W two K-steps ahead (global_prefetch_b8)
    if (ks + 2 < KSTEPS)
      __builtin_prefetch(Wg + (size_t)((ks + 2) * BK + b_kh * 16) * ND + (n0 + b_n), 0, 1);
  };

  // ---- registers -> bf16 LDS tiles --------------------------------------
  auto store_tile = [&](int buf) {
    #pragma unroll
    for (int j = 0; j < 4; ++j) {
      int f   = tid + 256 * j;
      int row = f >> 3;
      int kq  = (f & 7) * 4;
      u32x2 v;
      v.x = pack2_bf16(areg[j].x, areg[j].y);
      v.y = pack2_bf16(areg[j].z, areg[j].w);
      *(u32x2*)&As[buf][row * LDA + kq] = v;   // 8B store, 8B-aligned
    }
    u32x4 lo, hi;
    lo.x = pack2_bf16(breg[0],  breg[1]);  lo.y = pack2_bf16(breg[2],  breg[3]);
    lo.z = pack2_bf16(breg[4],  breg[5]);  lo.w = pack2_bf16(breg[6],  breg[7]);
    hi.x = pack2_bf16(breg[8],  breg[9]);  hi.y = pack2_bf16(breg[10], breg[11]);
    hi.z = pack2_bf16(breg[12], breg[13]); hi.w = pack2_bf16(breg[14], breg[15]);
    unsigned short* bd = &Bs[buf][b_n * LDB + b_kh * 16];  // transposed: [n][k]
    *(u32x4*)(bd)     = lo;                  // 16B stores, 16B-aligned
    *(u32x4*)(bd + 8) = hi;
  };

  v8f acc[4][2];
  {
    v8f z = {};
    #pragma unroll
    for (int a = 0; a < 4; ++a)
      #pragma unroll
      for (int b = 0; b < 2; ++b) acc[a][b] = z;
  }

  // ---- LDS -> fragments -> WMMA -----------------------------------------
  auto compute = [&](int buf) {
    v16bf afrag[4], bfrag[2];
    #pragma unroll
    for (int mt = 0; mt < 4; ++mt) {
      // A 16x32 bf16 layout: lane m = ln; e0..e7 -> K=8*lh..8*lh+7 ; e8..e15 -> +16
      const unsigned short* ap = &As[buf][(waveM + mt * 16 + ln) * LDA + lh * 8];
      u32x4 q0 = *(const u32x4*)(ap);        // ds_load_b128
      u32x4 q1 = *(const u32x4*)(ap + 16);   // +32 bytes
      u32x8 t;
      t[0] = q0.x; t[1] = q0.y; t[2] = q0.z; t[3] = q0.w;
      t[4] = q1.x; t[5] = q1.y; t[6] = q1.z; t[7] = q1.w;
      afrag[mt] = __builtin_bit_cast(v16bf, t);
    }
    #pragma unroll
    for (int nt = 0; nt < 2; ++nt) {
      // B 32x16 bf16 layout: lane n = ln; e0..e7 -> K=8*lh..8*lh+7 ; e8..e15 -> +16
      const unsigned short* bp = &Bs[buf][(waveN + nt * 16 + ln) * LDB + lh * 8];
      u32x4 q0 = *(const u32x4*)(bp);
      u32x4 q1 = *(const u32x4*)(bp + 16);
      u32x8 t;
      t[0] = q0.x; t[1] = q0.y; t[2] = q0.z; t[3] = q0.w;
      t[4] = q1.x; t[5] = q1.y; t[6] = q1.z; t[7] = q1.w;
      bfrag[nt] = __builtin_bit_cast(v16bf, t);
    }
    #pragma unroll
    for (int mt = 0; mt < 4; ++mt)
      #pragma unroll
      for (int nt = 0; nt < 2; ++nt)
        acc[mt][nt] = __builtin_amdgcn_wmma_f32_16x16x32_bf16(
            false, afrag[mt], false, bfrag[nt],
            (short)0, acc[mt][nt], false, false);
  };

  // ---- software-pipelined main loop (double-buffered LDS) ----------------
  load_tile(0);
  store_tile(0);
  __syncthreads();

  int p = 0;
  for (int ks = 0; ks < KSTEPS; ++ks) {
    if (ks + 1 < KSTEPS) load_tile(ks + 1);   // global loads overlap compute below
    compute(p);
    if (ks + 1 < KSTEPS) store_tile(p ^ 1);
    __syncthreads();
    p ^= 1;
  }

  // ---- store C (guard M edge) -------------------------------------------
  #pragma unroll
  for (int mt = 0; mt < 4; ++mt) {
    #pragma unroll
    for (int nt = 0; nt < 2; ++nt) {
      #pragma unroll
      for (int r = 0; r < 8; ++r) {
        // D layout: VGPR r holds M = r + 8*lh, N = ln
        int row = m0 + waveM + mt * 16 + lh * 8 + r;
        if (row < mg) {
          Out[(size_t)(row0 + row) * ND + (n0 + waveN + nt * 16 + ln)] = acc[mt][nt][r];
        }
      }
    }
  }
}

extern "C" void kernel_launch(void* const* d_in, const int* in_sizes, int n_in,
                              void* d_out, int out_size, void* d_ws, size_t ws_size,
                              hipStream_t stream) {
  const float* X = (const float*)d_in[0];
  const float* W = (const float*)d_in[1];
  float* Out = (float*)d_out;
  // Fixed per-expert token counts (module constants in the reference).
  static const int sizes[8] = {700, 300, 900, 150, 512, 600, 420, 514};
  int row0 = 0;
  for (int g = 0; g < 8; ++g) {
    const int mg = sizes[g];
    dim3 grid((mg + BM - 1) / BM, ND / BN, 1);
    grouped_gemm_wmma_bf16<<<grid, dim3(256, 1, 1), 0, stream>>>(
        X, W + (size_t)g * KD * ND, Out, row0, mg);
    row0 += mg;
  }
}
